// PIDLinear_56349970924056
// MI455X (gfx1250) — matrix operand order repaired and checked by reading, not compiled, not run on hardware
//
#include <hip/hip_runtime.h>
#include <hip/hip_bf16.h>

// ---------------- problem constants ----------------
#define BQ   4
#define TT   2048
#define DD   2048            // D_IN == D_OUT
#define RR   (BQ * TT)       // 8192 rows (tokens)
#define K3   (3 * DD)        // 6144 concatenated K
#define NN   DD              // 2048 output cols

// ---------------- vector types ----------------
typedef __attribute__((ext_vector_type(16))) __bf16          v16bf;
typedef __attribute__((ext_vector_type(8)))  float           v8f;
typedef __attribute__((ext_vector_type(8)))  unsigned short  v8u;
typedef __attribute__((ext_vector_type(4)))  unsigned int    v4u_t;
typedef __attribute__((ext_vector_type(8)))  int             v8i_t;
typedef __attribute__((ext_vector_type(4)))  int             v4i_t;

__device__ __forceinline__ unsigned short f2bf(float f) {
    union { float f; unsigned u; } v; v.f = f;
    unsigned r = v.u + 0x7FFFu + ((v.u >> 16) & 1u);   // round-to-nearest-even
    return (unsigned short)(r >> 16);
}
__device__ __forceinline__ float bf2f(unsigned short h) {
    union { unsigned u; float f; } v; v.u = ((unsigned)h) << 16; return v.f;
}
__device__ __forceinline__ v16bf mk16(v8u lo, v8u hi) {
    union { v16bf bf; v8u u[2]; } t; t.u[0] = lo; t.u[1] = hi; return t.bf;
}
// generic pointer to LDS: aperture lives in addr[63:32], offset in addr[31:0]
__device__ __forceinline__ unsigned lds_addr_of(const void* p) {
    return (unsigned)(unsigned long long)(uintptr_t)p;
}

// ============================================================
// Kernel 1: per-channel time scan -> bf16 concatenated features
// Acat[row, 0:D]=x, [D:2D]=integral, [2D:3D]=derivative (row = b*T+t)
// ============================================================
__global__ __launch_bounds__(256)
void scan_kernel(const float* __restrict__ x,
                 const float* __restrict__ alpha_logit,
                 const float* __restrict__ beta_logit,
                 unsigned short* __restrict__ Acat) {
    int c = blockIdx.x * 256 + threadIdx.x;   // 0 .. B*D-1
    int b = c >> 11;                           // /DD
    int d = c & (DD - 1);
    float alpha = 1.0f / (1.0f + __expf(-alpha_logit[d]));
    float beta  = 1.0f / (1.0f + __expf(-beta_logit[d]));
    float one_ma = 1.0f - alpha, one_mb = 1.0f - beta;

    const float* xp = x + (size_t)b * TT * DD + d;
    unsigned short* ap = Acat + (size_t)b * TT * K3 + d;

    float s = 0.0f, dv = 0.0f, xprev = 0.0f;
#pragma unroll 4
    for (int t = 0; t < TT; ++t) {
        float xt = xp[(size_t)t * DD];
        s  = alpha * s + one_ma * xt;
        dv = beta  * dv + one_mb * (xt - xprev);
        xprev = xt;
        size_t r = (size_t)t * K3;
        ap[r]          = f2bf(xt);
        ap[r + DD]     = f2bf(s);
        ap[r + 2 * DD] = f2bf(dv);
    }
}

// ============================================================
// Kernel 2: Wcat[o, 0:D]=W_p[o,:], [D:2D]=W_i, [2D:3D]=W_d  (bf16)
// ============================================================
__global__ __launch_bounds__(256)
void wcat_kernel(const float* __restrict__ Wp,
                 const float* __restrict__ Wi,
                 const float* __restrict__ Wd,
                 unsigned short* __restrict__ Wcat) {
    size_t i = (size_t)blockIdx.x * 256 + threadIdx.x;   // over N*K3
    size_t o = i / K3;
    int rem = (int)(i - o * K3);
    int f = rem >> 11;           // which segment
    int k = rem & (DD - 1);
    const float* W = (f == 0) ? Wp : ((f == 1) ? Wi : Wd);
    Wcat[i] = f2bf(W[o * DD + k]);
}

// ============================================================
// Kernel 3: gate logits + softmax, then scale Acat row in place.
// One 256-thread block per token row.
// ============================================================
__global__ __launch_bounds__(256)
void gates_kernel(unsigned short* __restrict__ Acat,
                  const float* __restrict__ Wg,    // (3, K3) row-major
                  const float* __restrict__ Wgb) { // (3,)
    int r = blockIdx.x;
    int tid = threadIdx.x;
    unsigned short* a = Acat + (size_t)r * K3;

    float s0 = 0.f, s1 = 0.f, s2 = 0.f;
    for (int k = tid; k < K3; k += 256) {
        float av = bf2f(a[k]);
        s0 += av * Wg[k];
        s1 += av * Wg[K3 + k];
        s2 += av * Wg[2 * K3 + k];
    }
    __shared__ float red0[256], red1[256], red2[256];
    red0[tid] = s0; red1[tid] = s1; red2[tid] = s2;
    __syncthreads();
    for (int off = 128; off > 0; off >>= 1) {
        if (tid < off) {
            red0[tid] += red0[tid + off];
            red1[tid] += red1[tid + off];
            red2[tid] += red2[tid + off];
        }
        __syncthreads();
    }
    __shared__ float g[3];
    if (tid == 0) {
        float l0 = red0[0] + Wgb[0];
        float l1 = red1[0] + Wgb[1];
        float l2 = red2[0] + Wgb[2];
        float m = fmaxf(l0, fmaxf(l1, l2));
        float e0 = __expf(l0 - m), e1 = __expf(l1 - m), e2 = __expf(l2 - m);
        float inv = 1.0f / (e0 + e1 + e2);
        g[0] = e0 * inv; g[1] = e1 * inv; g[2] = e2 * inv;
    }
    __syncthreads();
    float g0 = g[0], g1 = g[1], g2 = g[2];
    for (int k = tid; k < K3; k += 256) {
        int f = k >> 11;
        float gv = (f == 0) ? g0 : ((f == 1) ? g1 : g2);
        a[k] = f2bf(bf2f(a[k]) * gv);
    }
}

// ============================================================
// Kernel 4: y = Acat(8192 x 6144) @ Wcat(2048 x 6144)^T + bias
// 128x128 tile / block, BK=64, TDM double-buffered staging,
// 8 waves, each wave 64x32 (4x2 WMMA frags, 2 K-steps/tile)
// ============================================================
#define BM 128
#define BN 128
#define BK 64
#define LDA 72   // halves per LDS row = 64 + 8 pad (144 B, 16B-aligned, bank-spread)
#define NITER (K3 / BK)   // 96

// TDM group1 descriptor: shared by A and B tiles (same shape/stride).
// data_size=1 (2B), pad_enable, pad_interval=4 (32 DW = 128B row),
// pad_amount=3 (4 DW = 16B), tensor_dim0=6144, tensor_dim1=128,
// tile_dim0=64, tile_dim1=128, tensor_dim0_stride=6144.
__device__ __forceinline__ v8i_t tdm_group1() {
    v8i_t g1;
    g1[0] = (int)((1u << 16) | (1u << 20) | (4u << 22) | (3u << 25));
    g1[1] = (int)((unsigned)(K3 & 0xFFFF) << 16);   // tensor_dim0 lo16
    g1[2] = (int)((unsigned)BM << 16);              // tensor_dim0 hi16=0 | tensor_dim1 lo16
    g1[3] = (int)((unsigned)BK << 16);              // tensor_dim1 hi16=0 | tile_dim0
    g1[4] = (int)((unsigned)BM);                    // tile_dim1 | tile_dim2=0
    g1[5] = (int)((unsigned)K3);                    // tensor_dim0_stride lo32
    g1[6] = 0;                                      // stride hi16 | dim1_stride lo16
    g1[7] = 0;
    return g1;
}

__device__ __forceinline__ void tdm_load_tile(unsigned lds_byte_addr,
                                              const unsigned short* gptr,
                                              v8i_t g1) {
    unsigned long long ga = (unsigned long long)(uintptr_t)gptr;
    v4u_t g0;
    g0.x = 1u;                                        // count=1, user desc
    g0.y = lds_byte_addr;                             // lds_addr
    g0.z = (unsigned)(ga & 0xFFFFFFFFull);            // global_addr[31:0]
    g0.w = (unsigned)((ga >> 32) & 0x01FFFFFFull)     // global_addr[56:32]
         | 0x80000000u;                               // type=2 ("image")
    v4i_t z4 = (v4i_t)0;
    v8i_t z8 = (v8i_t)0;
    __builtin_amdgcn_tensor_load_to_lds(g0, g1, z4, z4, z8, 0);
}

__global__ __launch_bounds__(256)
void gemm_kernel(const unsigned short* __restrict__ Acat,
                 const unsigned short* __restrict__ Wcat,
                 const float* __restrict__ bias,
                 float* __restrict__ out) {
    __shared__ unsigned short As[2][BM * LDA];   // 2 x 18 KB
    __shared__ unsigned short Bs[2][BN * LDA];   // 2 x 18 KB   (72 KB total)

    const int tid  = threadIdx.x;
    const int wave = tid >> 5;        // 0..7
    const int lane = tid & 31;
    const int wm   = wave >> 2;       // 0..1 -> 64 rows each
    const int wn   = wave & 3;        // 0..3 -> 32 cols each
    const int r0   = blockIdx.x * BM; // M block
    const int n0   = blockIdx.y * BN; // N block

    const unsigned short* gA = Acat + (size_t)r0 * K3;   // tile base, advances by BK
    const unsigned short* gB = Wcat + (size_t)n0 * K3;

    v8f acc[4][2];
#pragma unroll
    for (int mi = 0; mi < 4; ++mi)
#pragma unroll
        for (int ni = 0; ni < 2; ++ni)
            acc[mi][ni] = (v8f)(0.0f);

    const v8i_t g1 = tdm_group1();
    const unsigned ldsA[2] = { lds_addr_of(&As[0][0]), lds_addr_of(&As[1][0]) };
    const unsigned ldsB[2] = { lds_addr_of(&Bs[0][0]), lds_addr_of(&Bs[1][0]) };

    // prologue: prefetch tile pairs 0 and 1
    if (wave == 0) {
        tdm_load_tile(ldsA[0], gA,      g1);
        tdm_load_tile(ldsB[0], gB,      g1);
        tdm_load_tile(ldsA[1], gA + BK, g1);
        tdm_load_tile(ldsB[1], gB + BK, g1);
    }

    // per-lane fragment addressing (ISA 7.12.2, 16-bit A 16x32 / B 32x16)
    const int mloc = lane & 15;
    const int kAlo = (lane >> 4) * 8;    // halves 0..7 -> K kAlo.. ; halves 8..15 -> +16
    const int kB   = (lane >> 4) * 8;    // halves 0..7 -> K kB.. ; halves 8..15 -> +16

    for (int it = 0; it < NITER; ++it) {
        if (wave == 0) {
            if (it + 1 < NITER) __builtin_amdgcn_s_wait_tensorcnt(2);
            else                __builtin_amdgcn_s_wait_tensorcnt(0);
        }
        __syncthreads();

        const int buf = it & 1;
        const unsigned short* as = &As[buf][0];
        const unsigned short* bs = &Bs[buf][0];
#pragma unroll
        for (int ks = 0; ks < 2; ++ks) {        // two K=32 steps within BK=64
            v16bf bfr[2];
#pragma unroll
            for (int ni = 0; ni < 2; ++ni) {
                int n = wn * 32 + ni * 16 + mloc;
                v8u lo = *(const v8u*)(bs + n * LDA + ks * 32 + kB);
                v8u hi = *(const v8u*)(bs + n * LDA + ks * 32 + kB + 16);
                bfr[ni] = mk16(lo, hi);
            }
#pragma unroll
            for (int mi = 0; mi < 4; ++mi) {
                int m = wm * 64 + mi * 16 + mloc;
                v8u lo = *(const v8u*)(as + m * LDA + ks * 32 + kAlo);
                v8u hi = *(const v8u*)(as + m * LDA + ks * 32 + kAlo + 16);
                v16bf afr = mk16(lo, hi);
#pragma unroll
                for (int ni = 0; ni < 2; ++ni) {
                    acc[mi][ni] = __builtin_amdgcn_wmma_f32_16x16x32_bf16(
                        false, afr, false, bfr[ni], (short)0, acc[mi][ni], false, false);
                }
            }
        }
        __syncthreads();

        // prefetch tile pair it+2 into the buffer just freed
        if (wave == 0 && it + 2 < NITER) {
            const unsigned short* a2 = gA + (size_t)(it + 2) * BK;
            const unsigned short* b2 = gB + (size_t)(it + 2) * BK;
            tdm_load_tile(ldsA[buf], a2, g1);
            tdm_load_tile(ldsB[buf], b2, g1);
        }
    }

    // epilogue: C layout — VGPR v: lanes 0-15 M=v, lanes 16-31 M=v+8; N=lane&15
    const int colb = n0 + wn * 32 + (lane & 15);
    const int rowh = (lane >> 4) * 8;
#pragma unroll
    for (int mi = 0; mi < 4; ++mi) {
#pragma unroll
        for (int ni = 0; ni < 2; ++ni) {
            int col = colb + ni * 16;
            float bv = bias[col];
            int rowbase = r0 + wm * 64 + mi * 16 + rowh;
#pragma unroll
            for (int v = 0; v < 8; ++v) {
                out[(size_t)(rowbase + v) * NN + col] = acc[mi][ni][v] + bv;
            }
        }
    }
}

// ============================================================
// launch
// ============================================================
extern "C" void kernel_launch(void* const* d_in, const int* in_sizes, int n_in,
                              void* d_out, int out_size, void* d_ws, size_t ws_size,
                              hipStream_t stream) {
    const float* x           = (const float*)d_in[0];
    const float* W_p         = (const float*)d_in[1];
    const float* W_i         = (const float*)d_in[2];
    const float* W_d         = (const float*)d_in[3];
    const float* alpha_logit = (const float*)d_in[4];
    const float* beta_logit  = (const float*)d_in[5];
    const float* W_gate_w    = (const float*)d_in[6];
    const float* W_gate_b    = (const float*)d_in[7];
    const float* bias        = (const float*)d_in[8];
    float* out = (float*)d_out;

    // workspace layout
    unsigned short* Acat = (unsigned short*)d_ws;     // RR*K3 bf16 = 96 MB
    unsigned short* Wcat = Acat + (size_t)RR * K3;    // NN*K3 bf16 = 24 MB

    // 1) scan -> bf16 concatenated features
    scan_kernel<<<(BQ * DD) / 256, 256, 0, stream>>>(x, alpha_logit, beta_logit, Acat);

    // 2) weights -> bf16 concatenated
    wcat_kernel<<<(int)(((size_t)NN * K3) / 256), 256, 0, stream>>>(W_p, W_i, W_d, Wcat);

    // 3) gates + in-place row scaling
    gates_kernel<<<RR, 256, 0, stream>>>(Acat, W_gate_w, W_gate_b);

    // 4) single fused GEMM (K = 6144) + bias, TDM double-buffered
    dim3 grid(RR / BM, NN / BN);
    gemm_kernel<<<grid, 256, 0, stream>>>(Acat, Wcat, bias, out);
}
